// MultiHeadSelfAttentionALiBi_9285719294396
// MI455X (gfx1250) — matrix operand order repaired and checked by reading, not compiled, not run on hardware
//
#include <hip/hip_runtime.h>
#include <math.h>

// ---------------------------------------------------------------------------
// Types for CDNA5 WMMA (wave32): D(16x16,f32) = A(16x32,bf16) x B(32x16,bf16) + C
// ---------------------------------------------------------------------------
typedef __bf16 bf16;
typedef bf16          v16bf __attribute__((ext_vector_type(16)));
typedef float         v8f   __attribute__((ext_vector_type(8)));
typedef unsigned int  v4u   __attribute__((ext_vector_type(4)));

#define B_   2
#define L_   2048
#define D_   768
#define H_   12
#define DH_  64
#define BL_  (B_ * L_)      // 4096
#define D3_  (3 * D_)       // 2304

static __device__ __forceinline__ v8f wmma_bf(v16bf a, v16bf b, v8f c) {
  // (neg_a, A, neg_b, B, c_mod, C, reuse_a, reuse_b)
  return __builtin_amdgcn_wmma_f32_16x16x32_bf16(false, a, false, b, (short)0, c,
                                                 false, false);
}

union FragU { v16bf v; v4u u[2]; };

// A-fragment (16x32 bf16, row-major source, leading dim ld):
// lane<16 holds row m=lane, K {0..7, 16..23}; lane>=16 holds K {8..15, 24..31}.
static __device__ __forceinline__ v16bf load_a_frag(const bf16* base, int ld, int lane) {
  int m = lane & 15, hf = lane >> 4;
  const bf16* p = base + m * ld + hf * 8;
  FragU f;
  f.u[0] = *(const v4u*)(p);
  f.u[1] = *(const v4u*)(p + 16);
  return f.v;
}

// B-fragment (32x16 bf16). Source is B^T row-major: bt[n][k], leading dim ldt.
// lane<16 holds column n=lane, K {0..7, 16..23}; lane>=16 holds K {8..15, 24..31}.
static __device__ __forceinline__ v16bf load_b_frag(const bf16* bt, int ldt, int lane) {
  int n = lane & 15, hf = lane >> 4;
  const bf16* p = bt + n * ldt + hf * 8;
  FragU f;
  f.u[0] = *(const v4u*)(p);
  f.u[1] = *(const v4u*)(p + 16);
  return f.v;
}

static __device__ __forceinline__ float alibi_slope(int h) {
  // H=12: first 8 heads 2^-(h+1); extra 4 heads 2^-((2*(h-8)+1)/2)
  return (h < 8) ? exp2f(-(float)(h + 1)) : exp2f(-0.5f * (float)(2 * (h - 8) + 1));
}

// ---------------------------------------------------------------------------
// Conversion kernels
// ---------------------------------------------------------------------------
__global__ void cvt_f32_bf16(const float* __restrict__ in, bf16* __restrict__ out, int n) {
  int i = blockIdx.x * blockDim.x + threadIdx.x;
  if (i < n) out[i] = (bf16)in[i];
}

// out is [C][R] bf16 = transpose of in [R][C] f32
__global__ void transpose_cvt(const float* __restrict__ in, bf16* __restrict__ out,
                              int R, int C) {
  int i = blockIdx.x * blockDim.x + threadIdx.x;
  if (i >= R * C) return;
  int c = i / R, r = i % R;
  out[i] = (bf16)in[r * C + c];
}

// ---------------------------------------------------------------------------
// QKV projection: qkv[4096,2304] = xbf @ Wqkv + bqkv, scattered into
//   Qbf [b*H+h][L][64], Kbf [b*H+h][L][64], Vt [b*H+h][64][L]   (all bf16)
// One wave computes a 16x64 strip: A-fragment reused across 4 B-fragments.
// All fragments for one K-chunk are loaded before the 4 WMMAs so the
// scheduler can overlap WMMA with in-flight loads (partial loadcnt waits).
// ---------------------------------------------------------------------------
__global__ __launch_bounds__(128) void qkv_gemm(const bf16* __restrict__ xbf,
                                                const bf16* __restrict__ wt,   // [2304][768]
                                                const float* __restrict__ bias,
                                                bf16* __restrict__ Qbf,
                                                bf16* __restrict__ Kbf,
                                                bf16* __restrict__ Vt) {
  int lane = threadIdx.x;
  int row0 = blockIdx.x * 16;
  int col0 = (blockIdx.y * 4 + threadIdx.y) * 64;   // 0..2240 step 64
  const bf16* a0 = xbf + (size_t)row0 * D_;
  const bf16* b0 = wt  + (size_t)col0 * D_;
  v8f acc[4] = {{}, {}, {}, {}};
#pragma unroll 2
  for (int k0 = 0; k0 < D_; k0 += 32) {
    v16bf a = load_a_frag(a0 + k0, D_, lane);
    v16bf bfr[4];
#pragma unroll
    for (int t = 0; t < 4; ++t)
      bfr[t] = load_b_frag(b0 + (size_t)t * 16 * D_ + k0, D_, lane);
#pragma unroll
    for (int t = 0; t < 4; ++t)
      acc[t] = wmma_bf(a, bfr[t], acc[t]);
  }

  int n = lane & 15, hf = lane >> 4;
#pragma unroll
  for (int t = 0; t < 4; ++t) {
    int col  = col0 + t * 16 + n;
    int part = col / D_, j = col % D_;
    int h = j >> 6, d = j & 63;
    float bv = bias[col];
#pragma unroll
    for (int r = 0; r < 8; ++r) {
      int row = row0 + r + 8 * hf;             // 0..4095
      int bb = row >> 11, q = row & (L_ - 1);
      bf16 o = (bf16)(acc[t][r] + bv);
      size_t bh = (size_t)(bb * H_ + h);
      if (part == 0)      Qbf[(bh * L_ + q) * DH_ + d] = o;
      else if (part == 1) Kbf[(bh * L_ + q) * DH_ + d] = o;
      else                Vt [(bh * DH_ + d) * L_ + q] = o;
    }
  }
}

// ---------------------------------------------------------------------------
// Flash-style attention with ALiBi + key mask. One wave per 16-query tile.
// S = Q K^T via 4 WMMAs per 32-key tile; online softmax (stats per-row in
// 16-lane halves via shfl_xor); P transposed through LDS to A-layout;
// O += P V via 4 WMMAs. Output bf16 [b][q][h*64+d].
// ---------------------------------------------------------------------------
__global__ __launch_bounds__(128) void attn_kernel(const bf16* __restrict__ Qbf,
                                                   const bf16* __restrict__ Kbf,
                                                   const bf16* __restrict__ Vt,
                                                   const int*  __restrict__ amask,
                                                   bf16* __restrict__ Obf) {
  __shared__ __align__(16) bf16 lds_p[4][16 * 32];
  int lane = threadIdx.x, w = threadIdx.y;
  int b = blockIdx.z, h = blockIdx.y;
  int q0 = (blockIdx.x * 4 + w) * 16;

  size_t bh = (size_t)(b * H_ + h);
  const bf16* Qp = Qbf + (bh * L_ + q0) * DH_;
  const bf16* Kp = Kbf + bh * L_ * DH_;
  const bf16* Vp = Vt  + bh * DH_ * L_;
  const int*  mk = amask + (size_t)b * L_;

  const float scale = 0.125f;                 // 1/sqrt(64)
  const float slope = alibi_slope(h);

  v16bf qa0 = load_a_frag(Qp,      DH_, lane);
  v16bf qa1 = load_a_frag(Qp + 32, DH_, lane);

  v8f o0 = {}, o1 = {}, o2 = {}, o3 = {};
  float m_i[8], l_i[8];
#pragma unroll
  for (int r = 0; r < 8; ++r) { m_i[r] = -__builtin_huge_valf(); l_i[r] = 0.0f; }

  int n = lane & 15, hf = lane >> 4;
  bf16* lp = &lds_p[w][0];

  for (int k0 = 0; k0 < L_; k0 += 32) {
    // ---- S tiles: 16q x 32k, reduction over dh=64 (two 32-chunks) ----
    // Load all four K fragments first so all 8 b128 loads are in flight
    // before the first WMMA (partial loadcnt waits instead of full drains).
    v16bf kb[4];
    kb[0] = load_b_frag(Kp + (size_t)k0 * DH_,             DH_, lane);
    kb[1] = load_b_frag(Kp + (size_t)k0 * DH_ + 32,        DH_, lane);
    kb[2] = load_b_frag(Kp + (size_t)(k0 + 16) * DH_,      DH_, lane);
    kb[3] = load_b_frag(Kp + (size_t)(k0 + 16) * DH_ + 32, DH_, lane);

    // Prefetch next key/value tiles toward the WGP (global_prefetch_b8);
    // softmax VALU work below hides the latency.
    if (k0 + 32 < L_) {
      __builtin_prefetch(Kp + (size_t)(k0 + 32) * DH_, 0, 3);
      __builtin_prefetch(Vp + (k0 + 32), 0, 3);
    }

    v8f s0 = {}, s1 = {};
    s0 = wmma_bf(qa0, kb[0], s0);
    s0 = wmma_bf(qa1, kb[1], s0);
    s1 = wmma_bf(qa0, kb[2], s1);
    s1 = wmma_bf(qa1, kb[3], s1);

    int kA = k0 + n, kB = kA + 16;
    bool okA = mk[kA] != 0, okB = mk[kB] != 0;

    float p0[8], p1[8];
#pragma unroll
    for (int r = 0; r < 8; ++r) {
      int qq = q0 + r + 8 * hf;
      float sA = okA ? (s0[r] * scale - slope * fabsf((float)(qq - kA))) : -1.0e9f;
      float sB = okB ? (s1[r] * scale - slope * fabsf((float)(qq - kB))) : -1.0e9f;
      // row-max over 32 keys: local pair-max then butterfly over 16-lane half
      float t = fmaxf(sA, sB);
      t = fmaxf(t, __shfl_xor(t, 1, 32));
      t = fmaxf(t, __shfl_xor(t, 2, 32));
      t = fmaxf(t, __shfl_xor(t, 4, 32));
      t = fmaxf(t, __shfl_xor(t, 8, 32));
      float mnew  = fmaxf(m_i[r], t);
      float alpha = __expf(m_i[r] - mnew);
      float pA = __expf(sA - mnew);
      float pB = __expf(sB - mnew);
      float rs = pA + pB;
      rs += __shfl_xor(rs, 1, 32);
      rs += __shfl_xor(rs, 2, 32);
      rs += __shfl_xor(rs, 4, 32);
      rs += __shfl_xor(rs, 8, 32);
      l_i[r] = l_i[r] * alpha + rs;
      m_i[r] = mnew;
      o0[r] *= alpha; o1[r] *= alpha; o2[r] *= alpha; o3[r] *= alpha;
      p0[r] = pA; p1[r] = pB;
    }

    // ---- transpose P (D-layout f32) -> A-layout bf16 via LDS ----
#pragma unroll
    for (int r = 0; r < 8; ++r) {
      int rr = r + 8 * hf;
      lp[rr * 32 + n]      = (bf16)p0[r];
      lp[rr * 32 + 16 + n] = (bf16)p1[r];
    }

    // Issue all four V-fragment loads alongside the LDS readback of P.
    v16bf vb[4];
    vb[0] = load_b_frag(Vp +           k0, L_, lane);
    vb[1] = load_b_frag(Vp + 16 * L_ + k0, L_, lane);
    vb[2] = load_b_frag(Vp + 32 * L_ + k0, L_, lane);
    vb[3] = load_b_frag(Vp + 48 * L_ + k0, L_, lane);
    v16bf pa = load_a_frag(lp, 32, lane);   // compiler inserts s_wait_dscnt

    // ---- O += P * V  (V^T rows are d, columns are keys; ldt = L) ----
    o0 = wmma_bf(pa, vb[0], o0);
    o1 = wmma_bf(pa, vb[1], o1);
    o2 = wmma_bf(pa, vb[2], o2);
    o3 = wmma_bf(pa, vb[3], o3);
  }

  // ---- epilogue: normalize and store bf16 [b][q][h*64 + d] ----
#pragma unroll
  for (int r = 0; r < 8; ++r) {
    float inv = 1.0f / l_i[r];
    int rr = r + 8 * hf;
    size_t base = ((size_t)b * L_ + (q0 + rr)) * D_ + h * DH_ + n;
    Obf[base]      = (bf16)(o0[r] * inv);
    Obf[base + 16] = (bf16)(o1[r] * inv);
    Obf[base + 32] = (bf16)(o2[r] * inv);
    Obf[base + 48] = (bf16)(o3[r] * inv);
  }
}

// ---------------------------------------------------------------------------
// Output projection: out[4096,768] (f32) = Obf @ Wout + bout
// One wave computes a 16x64 strip: A-fragment reused across 4 B-fragments.
// ---------------------------------------------------------------------------
__global__ __launch_bounds__(128) void out_gemm(const bf16* __restrict__ a,
                                                const bf16* __restrict__ wt,  // [768][768] = Wout^T
                                                const float* __restrict__ bias,
                                                float* __restrict__ out) {
  int lane = threadIdx.x;
  int row0 = blockIdx.x * 16;
  int col0 = (blockIdx.y * 4 + threadIdx.y) * 64;   // 0..704 step 64
  const bf16* a0 = a  + (size_t)row0 * D_;
  const bf16* b0 = wt + (size_t)col0 * D_;
  v8f acc[4] = {{}, {}, {}, {}};
#pragma unroll 2
  for (int k0 = 0; k0 < D_; k0 += 32) {
    v16bf av = load_a_frag(a0 + k0, D_, lane);
    v16bf bfr[4];
#pragma unroll
    for (int t = 0; t < 4; ++t)
      bfr[t] = load_b_frag(b0 + (size_t)t * 16 * D_ + k0, D_, lane);
#pragma unroll
    for (int t = 0; t < 4; ++t)
      acc[t] = wmma_bf(av, bfr[t], acc[t]);
  }

  int n = lane & 15, hf = lane >> 4;
#pragma unroll
  for (int t = 0; t < 4; ++t) {
    int col = col0 + t * 16 + n;
    float bv = bias[col];
#pragma unroll
    for (int r = 0; r < 8; ++r)
      out[(size_t)(row0 + r + 8 * hf) * D_ + col] = acc[t][r] + bv;
  }
}

// ---------------------------------------------------------------------------
extern "C" void kernel_launch(void* const* d_in, const int* in_sizes, int n_in,
                              void* d_out, int out_size, void* d_ws, size_t ws_size,
                              hipStream_t stream) {
  (void)in_sizes; (void)n_in; (void)out_size; (void)ws_size;
  const float* x     = (const float*)d_in[0];
  const int*   amask = (const int*)  d_in[1];
  const float* Wqkv  = (const float*)d_in[2];
  const float* bqkv  = (const float*)d_in[3];
  const float* Wout  = (const float*)d_in[4];
  const float* bout  = (const float*)d_in[5];
  float* out = (float*)d_out;

  // workspace carving (bf16 elements)
  bf16* xbf   = (bf16*)d_ws;                       // 4096*768
  bf16* wqkvT = xbf   + (size_t)BL_ * D_;          // 2304*768
  bf16* woutT = wqkvT + (size_t)D3_ * D_;          // 768*768
  bf16* qbf   = woutT + (size_t)D_  * D_;          // 2*12*2048*64
  bf16* kbf   = qbf   + (size_t)B_ * H_ * L_ * DH_;
  bf16* vt    = kbf   + (size_t)B_ * H_ * L_ * DH_;
  bf16* obf   = vt    + (size_t)B_ * H_ * L_ * DH_; // 4096*768

  const int nx  = BL_ * D_;                 // 3145728
  const int nwq = D_ * D3_;                 // 1769472
  const int nwo = D_ * D_;                  // 589824

  cvt_f32_bf16 <<<(nx  + 255) / 256, 256, 0, stream>>>(x, xbf, nx);
  transpose_cvt<<<(nwq + 255) / 256, 256, 0, stream>>>(Wqkv, wqkvT, D_, D3_);
  transpose_cvt<<<(nwo + 255) / 256, 256, 0, stream>>>(Wout, woutT, D_, D_);

  // 16x64 strip per wave, 4 waves (256 cols) per block
  qkv_gemm   <<<dim3(BL_ / 16, D3_ / 256), dim3(32, 4), 0, stream>>>(
      xbf, wqkvT, bqkv, qbf, kbf, vt);

  attn_kernel<<<dim3(L_ / 16 / 4, H_, B_), dim3(32, 4), 0, stream>>>(
      qbf, kbf, vt, amask, obf);

  out_gemm   <<<dim3(BL_ / 16, D_ / 256), dim3(32, 4), 0, stream>>>(
      obf, woutT, bout, out);
}
